// MoEFused_53025666236535
// MI455X (gfx1250) — matrix operand order repaired and checked by reading, not compiled, not run on hardware
//
#include <hip/hip_runtime.h>
#include <stdint.h>

#define T_TOK   8192
#define DMODEL  1024
#define FFN     4096
#define NEXP    8
#define LDH     (FFN + 8)          // padded sh_h row stride (elements): conflict-free A reads
#define MAXTILE 512                // T_TOK / 16
#define NB      4                  // N-tile register blocking per wave

typedef __attribute__((ext_vector_type(16))) __bf16 v16bf;
typedef __attribute__((ext_vector_type(8)))  float  v8f;
typedef __attribute__((ext_vector_type(4)))  unsigned int u32x4;
typedef __attribute__((ext_vector_type(8)))  int i32x8;
typedef __attribute__((ext_vector_type(4)))  int i32x4;

// float -> bf16 with round-to-nearest-even
static __device__ __forceinline__ unsigned short f2bf(float f) {
  unsigned u = __float_as_uint(f);
  u += 0x7FFFu + ((u >> 16) & 1u);
  return (unsigned short)(u >> 16);
}

// sh_x padded layout: 16 B of pad after every 1024 data bytes (matches TDM pad_interval=1KB,
// pad_amount=4 dwords). Row stride 2080 B -> rows 8 banks apart (2-way max conflict).
static __device__ __forceinline__ int xpad(int m, int k /*element*/) {
  int L = (m * DMODEL + k) * 2;
  return L + ((L >> 10) << 4);
}

// A fragment (16x32 bf16) from padded sh_x; lanes 0-15: K 0..15, lanes 16-31: K 16..31.
static __device__ __forceinline__ v16bf load_a_x(const char* shx, int k0, int lane) {
  int m = lane & 15, hi = (lane >> 4) << 4;
  union { v16bf v; uint4 q[2]; } u;
  u.q[0] = *(const uint4*)(shx + xpad(m, k0 + hi));
  u.q[1] = *(const uint4*)(shx + xpad(m, k0 + hi + 8));
  return u.v;
}

// A fragment from sh_h (row stride LDH elements, 16B-aligned chunks, banks 4 apart).
static __device__ __forceinline__ v16bf load_a_h(const unsigned short* shh, int k0, int lane) {
  int m = lane & 15, hi = (lane >> 4) << 4;
  const unsigned short* p = shh + m * LDH + k0 + hi;
  union { v16bf v; uint4 q[2]; } u;
  u.q[0] = *(const uint4*)p;
  u.q[1] = *(const uint4*)(p + 8);
  return u.v;
}

// B fragment (32x16) from row-major f32 weights [K x N], cvt to bf16 on the fly (slow path).
static __device__ __forceinline__ v16bf load_b_f32(const float* __restrict__ B, int ldb,
                                                   int k0, int n0, int lane) {
  int n  = n0 + (lane & 15);
  int ks = k0 + ((lane >> 4) << 4);
  const float* p = B + (size_t)ks * ldb + n;
  union { v16bf v; unsigned short s[16]; } u;
#pragma unroll
  for (int i = 0; i < 16; ++i) u.s[i] = f2bf(p[(size_t)i * ldb]);
  return u.v;
}

// B fragment from pre-swizzled bf16 weights: layout [nt][ks][lane][16 elems] -> two b128 loads.
static __device__ __forceinline__ v16bf load_b_swz(const unsigned short* __restrict__ B,
                                                   int K32, int nt, int ks, int lane) {
  const unsigned short* p = B + ((((size_t)nt * K32 + ks) * 32 + lane) << 4);
  union { v16bf v; uint4 q[2]; } u;
  u.q[0] = *(const uint4*)p;
  u.q[1] = *(const uint4*)(p + 8);
  return u.v;
}

static __device__ __forceinline__ v8f wmma_bf16(v16bf a, v16bf b, v8f c) {
  return __builtin_amdgcn_wmma_f32_16x16x32_bf16(false, a, false, b, (short)0, c, false, false);
}

// ---------------- Kernel 0: f32 [K x N] -> bf16 WMMA-native swizzled fragments --------------
__global__ __launch_bounds__(256) void k_swizzle(const float* __restrict__ src,
                                                 unsigned short* __restrict__ dst,
                                                 int K, int N) {
  const int e = blockIdx.y;
  const float* s = src + (size_t)e * K * N;
  unsigned short* d = dst + (size_t)e * K * N;
  const int K32 = K >> 5, NT = N >> 4;
  const int total = NT * K32 * 32;
  for (int idx = blockIdx.x * blockDim.x + threadIdx.x; idx < total;
       idx += gridDim.x * blockDim.x) {
    int nt   = idx / (K32 * 32);
    int rem  = idx - nt * K32 * 32;
    int ks   = rem >> 5, lane = rem & 31;
    int n    = nt * 16 + (lane & 15);
    int k0   = ks * 32 + ((lane >> 4) << 4);
    union { uint4 q[2]; unsigned short sh[16]; } u;
#pragma unroll
    for (int i = 0; i < 16; ++i) u.sh[i] = f2bf(s[(size_t)(k0 + i) * N + n]);
    uint4* dp = (uint4*)(d + ((size_t)idx << 4));
    dp[0] = u.q[0]; dp[1] = u.q[1];
  }
}

// ---------------- Kernel 1: RMSNorm + router + top-2 + lists + zero-init out ---------------
__global__ __launch_bounds__(256) void k_router(
    const float* __restrict__ x, const float* __restrict__ lnw,
    const float* __restrict__ wr,
    unsigned short* __restrict__ xnb, float* __restrict__ out,
    float* __restrict__ logits, int* __restrict__ cnt,
    int* __restrict__ tlist, float* __restrict__ wlist)
{
  int lane = threadIdx.x & 31;
  int wave = threadIdx.x >> 5;
  int t = blockIdx.x * 8 + wave;
  if (t >= T_TOK) return;
  const float* xr = x + (size_t)t * DMODEL;

  float v[32]; float ss = 0.f;
#pragma unroll
  for (int i = 0; i < 32; ++i) { float f = xr[i * 32 + lane]; v[i] = f; ss += f * f; }
#pragma unroll
  for (int o = 16; o > 0; o >>= 1) ss += __shfl_xor(ss, o, 32);
  float rstd = rsqrtf(ss * (1.f / DMODEL) + 1e-6f);

  float lg[NEXP];
#pragma unroll
  for (int e = 0; e < NEXP; ++e) lg[e] = 0.f;
#pragma unroll
  for (int i = 0; i < 32; ++i) {
    int idx = i * 32 + lane;
    float xv = v[i] * rstd * lnw[idx];
    xnb[(size_t)t * DMODEL + idx] = f2bf(xv);
    out[(size_t)t * DMODEL + idx] = 0.f;             // zero-init output accumulator
    const float* wrow = wr + idx * NEXP;
#pragma unroll
    for (int e = 0; e < NEXP; ++e) lg[e] = __fmaf_rn(xv, wrow[e], lg[e]);
  }
#pragma unroll
  for (int e = 0; e < NEXP; ++e) {
#pragma unroll
    for (int o = 16; o > 0; o >>= 1) lg[e] += __shfl_xor(lg[e], o, 32);
  }

  if (lane == 0) {
    int i1 = 0; float m1 = lg[0];
#pragma unroll
    for (int e = 1; e < NEXP; ++e) if (lg[e] > m1) { m1 = lg[e]; i1 = e; }
    int i2 = (i1 == 0) ? 1 : 0; float m2 = lg[i2];
#pragma unroll
    for (int e = 0; e < NEXP; ++e) if (e != i1 && lg[e] > m2) { m2 = lg[e]; i2 = e; }
    // renormalized top-2 softmax weights: p1/(p1+p2) = sigmoid(l1-l2)
    float w1 = 1.f / (1.f + __expf(m2 - m1));
    float w2 = 1.f - w1;
#pragma unroll
    for (int e = 0; e < NEXP; ++e) logits[(size_t)t * NEXP + e] = lg[e];
    int p1 = atomicAdd(&cnt[i1], 1);
    tlist[i1 * T_TOK + p1] = t; wlist[i1 * T_TOK + p1] = w1;
    int p2 = atomicAdd(&cnt[i2], 1);
    tlist[i2 * T_TOK + p2] = t; wlist[i2 * T_TOK + p2] = w2;
  }
}

// ---------------- Kernel 2: per-expert SwiGLU FFN on gathered 16-token tiles ----------------
template <bool SWZ>
__global__ __launch_bounds__(512, 1) void k_expert(
    const unsigned short* __restrict__ xnb,
    const float* __restrict__ wg, const float* __restrict__ wu,
    const float* __restrict__ wd,
    const unsigned short* __restrict__ wgs, const unsigned short* __restrict__ wus,
    const unsigned short* __restrict__ wds,
    const int* __restrict__ cnt, const int* __restrict__ tlist,
    const float* __restrict__ wlist, float* __restrict__ out)
{
  __shared__ __align__(16) char           sh_x[16 * (2048 + 32)];  // padded xn tile (bf16)
  __shared__ __align__(16) unsigned short sh_h[16 * LDH];          // h tile (bf16)
  __shared__ int   sh_tok[16];
  __shared__ float sh_w[16];

  const int e     = blockIdx.x >> 9;
  const int tile  = blockIdx.x & (MAXTILE - 1);
  const int count = cnt[e];
  if (tile * 16 >= count) return;

  const int tid = threadIdx.x, lane = tid & 31, wave = tid >> 5;

  if (tid < 16) {
    int slot  = tile * 16 + tid;
    int t0    = tlist[e * T_TOK + tile * 16];
    bool ok   = slot < count;
    sh_tok[tid] = ok ? tlist[e * T_TOK + slot] : t0;  // pad rows re-load a valid token
    sh_w[tid]   = ok ? wlist[e * T_TOK + slot] : 0.f; // ...but contribute exactly 0
  }
  __syncthreads();

  // --- Stage gathered token rows into LDS via Tensor Data Mover (gather mode) ---
#if __has_builtin(__builtin_amdgcn_tensor_load_to_lds)
  if (wave == 0) {
    uint64_t ga   = (uint64_t)(uintptr_t)xnb;
    uint32_t ldsa = (uint32_t)(uintptr_t)(&sh_x[0]);
    u32x4 g0;
    g0[0] = 0x80000001u;                         // count=1, gather_mode=1, 16-bit indices
    g0[1] = ldsa;                                // LDS dest
    g0[2] = (uint32_t)ga;                        // global_addr lo
    g0[3] = (uint32_t)(ga >> 32) | (2u << 30);   // global_addr hi | type=2 (image)
    i32x8 g1;
    g1[0] = (1 << 16)                            // data_size = 2 B
          | (1 << 20)                            // pad_enable
          | (7 << 22)                            // pad_interval = 256 dwords (1 KB)
          | (3 << 25);                           // pad_amount   = 4 dwords (16 B)
    g1[1] = (DMODEL & 0xFFFF) << 16;                                // tensor_dim0 lo
    g1[2] = ((DMODEL >> 16) & 0xFFFF) | ((T_TOK & 0xFFFF) << 16);   // dim0 hi | dim1 lo
    g1[3] = ((T_TOK >> 16) & 0xFFFF) | (DMODEL << 16);              // dim1 hi | tile_dim0
    g1[4] = 16;                                  // tile_dim1 = #valid gather indices
    g1[5] = DMODEL;                              // tensor_dim0_stride lo32
    g1[6] = 0;
    g1[7] = 0;
    i32x4 g2, g3;
#pragma unroll
    for (int j = 0; j < 4; ++j) {
      g2[j] = (sh_tok[2 * j] & 0xFFFF)     | (sh_tok[2 * j + 1] << 16);
      g3[j] = (sh_tok[8 + 2 * j] & 0xFFFF) | (sh_tok[8 + 2 * j + 1] << 16);
    }
#if defined(__clang_major__) && (__clang_major__ >= 23)
    i32x8 g4 = {0, 0, 0, 0, 0, 0, 0, 0};
    __builtin_amdgcn_tensor_load_to_lds(g0, g1, g2, g3, g4, 0);
#else
    __builtin_amdgcn_tensor_load_to_lds(g0, g1, g2, g3, 0);
#endif
    __builtin_amdgcn_s_wait_tensorcnt(0);
  }
#else
  for (int m = wave; m < 16; m += 16) {
    const unsigned short* src = xnb + (size_t)sh_tok[m] * DMODEL;
    for (int k = lane * 8; k < DMODEL; k += 32 * 8) {
      uint4 d = *(const uint4*)(src + k);
      *(uint4*)(sh_x + xpad(m, k)) = d;
    }
  }
#endif
  __syncthreads();

  // --- Phase 1: h = silu(xn @ Wg) * (xn @ Wu),  M=16 N=4096 K=1024, bf16 WMMA ---
  // NB-blocked: one LDS A fragment feeds 2*NB WMMAs; k-loop kept rolled (unroll 4) so
  // A loads stay in-loop (prevents LICM hoist -> scratch spill).
  const float* wge = wg + (size_t)e * DMODEL * FFN;
  const float* wue = wu + (size_t)e * DMODEL * FFN;
  const unsigned short* wgse = wgs + (size_t)e * DMODEL * FFN;
  const unsigned short* wuse = wus + (size_t)e * DMODEL * FFN;
  for (int ntb = wave * NB; ntb < FFN / 16; ntb += 16 * NB) {
    v8f cg[NB], cu[NB];
#pragma unroll
    for (int j = 0; j < NB; ++j) {
      cg[j] = (v8f){0, 0, 0, 0, 0, 0, 0, 0};
      cu[j] = (v8f){0, 0, 0, 0, 0, 0, 0, 0};
    }
#pragma unroll 4
    for (int ks = 0; ks < DMODEL / 32; ++ks) {
      v16bf a = load_a_x(sh_x, ks * 32, lane);
#pragma unroll
      for (int j = 0; j < NB; ++j) {
        v16bf bg, bu;
        if (SWZ) {
          bg = load_b_swz(wgse, DMODEL / 32, ntb + j, ks, lane);
          bu = load_b_swz(wuse, DMODEL / 32, ntb + j, ks, lane);
        } else {
          bg = load_b_f32(wge, FFN, ks * 32, (ntb + j) * 16, lane);
          bu = load_b_f32(wue, FFN, ks * 32, (ntb + j) * 16, lane);
        }
        cg[j] = wmma_bf16(a, bg, cg[j]);
        cu[j] = wmma_bf16(a, bu, cu[j]);
      }
    }
    int mbase = (lane >> 4) << 3;
#pragma unroll
    for (int j = 0; j < NB; ++j) {
      int n = (ntb + j) * 16 + (lane & 15);
#pragma unroll
      for (int r = 0; r < 8; ++r) {
        float g = cg[j][r], u = cu[j][r];
        // silu(g)*u with fast v_rcp instead of IEEE divide
        float h = g * __builtin_amdgcn_rcpf(1.f + __expf(-g)) * u;
        sh_h[(mbase + r) * LDH + n] = f2bf(h);
      }
    }
  }
  __syncthreads();

  // --- Phase 2: out += w * (h @ Wd),  M=16 N=1024 K=4096, bf16 WMMA + f32 atomics ---
  const float* wde = wd + (size_t)e * FFN * DMODEL;
  const unsigned short* wdse = wds + (size_t)e * FFN * DMODEL;
  for (int ntb = wave * NB; ntb < DMODEL / 16; ntb += 16 * NB) {
    v8f c[NB];
#pragma unroll
    for (int j = 0; j < NB; ++j) c[j] = (v8f){0, 0, 0, 0, 0, 0, 0, 0};
#pragma unroll 4
    for (int ks = 0; ks < FFN / 32; ++ks) {
      v16bf a = load_a_h(sh_h, ks * 32, lane);
#pragma unroll
      for (int j = 0; j < NB; ++j) {
        v16bf b;
        if (SWZ) b = load_b_swz(wdse, FFN / 32, ntb + j, ks, lane);
        else     b = load_b_f32(wde, DMODEL, ks * 32, (ntb + j) * 16, lane);
        c[j] = wmma_bf16(a, b, c[j]);
      }
    }
    int mbase = (lane >> 4) << 3;
#pragma unroll
    for (int j = 0; j < NB; ++j) {
      int n = (ntb + j) * 16 + (lane & 15);
#pragma unroll
      for (int r = 0; r < 8; ++r) {
        float vv = c[j][r] * sh_w[mbase + r];
        atomicAdd(out + (size_t)sh_tok[mbase + r] * DMODEL + n, vv);
      }
    }
  }
}

// ------------------------------------ Launcher ------------------------------------
extern "C" void kernel_launch(void* const* d_in, const int* in_sizes, int n_in,
                              void* d_out, int out_size, void* d_ws, size_t ws_size,
                              hipStream_t stream) {
  const float* x   = (const float*)d_in[0];
  const float* lnw = (const float*)d_in[1];
  const float* wr  = (const float*)d_in[2];
  const float* wg  = (const float*)d_in[3];
  const float* wu  = (const float*)d_in[4];
  const float* wd  = (const float*)d_in[5];
  float* out    = (float*)d_out;
  float* logits = out + (size_t)T_TOK * DMODEL;

  char* ws = (char*)d_ws;
  unsigned short* xnb = (unsigned short*)ws;                     // 16 MB: xn in bf16
  size_t off = (size_t)T_TOK * DMODEL * sizeof(unsigned short);
  int* cnt = (int*)(ws + off); off += 256;                       // per-expert counts
  int* tlist = (int*)(ws + off); off += (size_t)NEXP * T_TOK * sizeof(int);
  float* wlist = (float*)(ws + off); off += (size_t)NEXP * T_TOK * sizeof(float);

  // Optional pre-swizzled bf16 weight region (3 x 64 MB)
  const size_t wbytes = (size_t)NEXP * DMODEL * FFN * sizeof(unsigned short);
  off = (off + 255) & ~(size_t)255;
  unsigned short* wgs = (unsigned short*)(ws + off);
  unsigned short* wus = (unsigned short*)(ws + off + wbytes);
  unsigned short* wds = (unsigned short*)(ws + off + 2 * wbytes);
  const bool swz = ws_size >= off + 3 * wbytes;

  hipMemsetAsync(cnt, 0, 256, stream);
  k_router<<<T_TOK / 8, 256, 0, stream>>>(x, lnw, wr, xnb, out, logits, cnt, tlist, wlist);

  if (swz) {
    k_swizzle<<<dim3(1024, NEXP), 256, 0, stream>>>(wg, wgs, DMODEL, FFN);
    k_swizzle<<<dim3(1024, NEXP), 256, 0, stream>>>(wu, wus, DMODEL, FFN);
    k_swizzle<<<dim3(1024, NEXP), 256, 0, stream>>>(wd, wds, FFN, DMODEL);
    k_expert<true><<<NEXP * MAXTILE, 512, 0, stream>>>(
        xnb, wg, wu, wd, wgs, wus, wds, cnt, tlist, wlist, out);
  } else {
    k_expert<false><<<NEXP * MAXTILE, 512, 0, stream>>>(
        xnb, wg, wu, wd, xnb, xnb, xnb, cnt, tlist, wlist, out);
  }

  (void)in_sizes; (void)n_in; (void)out_size; (void)ws_size;
}
